// HierarchicalVNN_37795712205050
// MI455X (gfx1250) — matrix operand order repaired and checked by reading, not compiled, not run on hardware
//
#include <hip/hip_runtime.h>

typedef __attribute__((ext_vector_type(16))) _Float16 v16h;
typedef __attribute__((ext_vector_type(8)))  _Float16 v8h;
typedef __attribute__((ext_vector_type(8)))  float    v8f;

#define BATCH 4
#define NPTS  2048
#define KNN   20
#define MCOL  (BATCH * 3 * NPTS)   // 24576 columns = (b, vec, n)
#define EPSV  1e-6f

// ---------------------------------------------------------------------------
// 1. kNN: one wave32 per query point; 20 max-selection passes with the same
//    tie-breaking as jax.lax.top_k (ties -> smaller index first).
// ---------------------------------------------------------------------------
__global__ __launch_bounds__(256)
void knn_kernel(const float* __restrict__ p, int* __restrict__ idx)
{
    __shared__ float sx[NPTS], sy[NPTS], sz[NPTS], sd2[NPTS];
    const int b   = blockIdx.y;
    const int tid = threadIdx.x;
    for (int m = tid; m < NPTS; m += 256) {
        const float x = p[((size_t)b * NPTS + m) * 3 + 0];
        const float y = p[((size_t)b * NPTS + m) * 3 + 1];
        const float z = p[((size_t)b * NPTS + m) * 3 + 2];
        sx[m] = x; sy[m] = y; sz[m] = z; sd2[m] = x * x + y * y + z * z;
    }
    __syncthreads();

    const int lane = tid & 31;
    const int n    = blockIdx.x * 8 + (tid >> 5);     // 8 waves -> 8 queries
    const float qx = sx[n], qy = sy[n], qz = sz[n], q2 = sd2[n];

    float last_s = 3.4e38f;
    int   last_i = -1;
    for (int j = 0; j < KNN; ++j) {
        float best = -3.4e38f;
        int   bi   = 0x7fffffff;
        for (int c = 0; c < NPTS / 32; ++c) {
            const int   m = c * 32 + lane;
            const float s = 2.f * (qx * sx[m] + qy * sy[m] + qz * sz[m]) - q2 - sd2[m];
            const bool elig = (s < last_s) || (s == last_s && m > last_i);
            if (elig && (s > best || (s == best && m < bi))) { best = s; bi = m; }
        }
        #pragma unroll
        for (int off = 16; off > 0; off >>= 1) {       // wave32 xor-reduce
            const float os = __shfl_xor(best, off, 32);
            const int   oi = __shfl_xor(bi,   off, 32);
            if (os > best || (os == best && oi < bi)) { best = os; bi = oi; }
        }
        if (lane == 0) idx[((size_t)b * NPTS + n) * KNN + j] = bi;
        last_s = best; last_i = bi;
    }
}

// ---------------------------------------------------------------------------
// 2. Edge features (rel / center / cross) + VN conv (C=3) + leaky(0.2) + mean
//    over k.  Output X0 laid out as [128 rows x MCOL cols], col=(b*3+v)*N+n.
// ---------------------------------------------------------------------------
__global__ __launch_bounds__(128)
void edge_conv_kernel(const float* __restrict__ p, const int* __restrict__ idx,
                      const float* __restrict__ wf, const float* __restrict__ wd,
                      float* __restrict__ X0)
{
    __shared__ float feat[KNN][3][3];   // [k][channel: rel,ctr,cross][vec]
    const int b = blockIdx.y, n = blockIdx.x, tid = threadIdx.x;

    if (tid < KNN) {
        const int j = idx[((size_t)b * NPTS + n) * KNN + tid];
        const float nx = p[((size_t)b * NPTS + j) * 3 + 0];
        const float ny = p[((size_t)b * NPTS + j) * 3 + 1];
        const float nz = p[((size_t)b * NPTS + j) * 3 + 2];
        const float cx = p[((size_t)b * NPTS + n) * 3 + 0];
        const float cy = p[((size_t)b * NPTS + n) * 3 + 1];
        const float cz = p[((size_t)b * NPTS + n) * 3 + 2];
        feat[tid][0][0] = nx - cx; feat[tid][0][1] = ny - cy; feat[tid][0][2] = nz - cz;
        feat[tid][1][0] = cx;      feat[tid][1][1] = cy;      feat[tid][1][2] = cz;
        feat[tid][2][0] = ny * cz - nz * cy;                  // cross(nb, ctr)
        feat[tid][2][1] = nz * cx - nx * cz;
        feat[tid][2][2] = nx * cy - ny * cx;
    }
    __syncthreads();

    const int o = tid;
    const float wf0 = wf[o * 3 + 0], wf1 = wf[o * 3 + 1], wf2 = wf[o * 3 + 2];
    const float wd0 = wd[o * 3 + 0], wd1 = wd[o * 3 + 1], wd2 = wd[o * 3 + 2];
    float acc[3] = {0.f, 0.f, 0.f};
    for (int k = 0; k < KNN; ++k) {
        float nv[3], dv[3];
        #pragma unroll
        for (int v = 0; v < 3; ++v) {
            nv[v] = wf0 * feat[k][0][v] + wf1 * feat[k][1][v] + wf2 * feat[k][2][v];
            dv[v] = wd0 * feat[k][0][v] + wd1 * feat[k][1][v] + wd2 * feat[k][2][v];
        }
        const float dot = nv[0] * dv[0] + nv[1] * dv[1] + nv[2] * dv[2];
        const float dsq = dv[0] * dv[0] + dv[1] * dv[1] + dv[2] * dv[2];
        const float t   = dot / (dsq + EPSV);
        #pragma unroll
        for (int v = 0; v < 3; ++v) {
            const float xv = nv[v];
            const float xn = xv - t * dv[v];
            acc[v] += 0.2f * xv + 0.8f * ((dot >= 0.f) ? xv : xn);
        }
    }
    #pragma unroll
    for (int v = 0; v < 3; ++v)
        X0[(size_t)o * MCOL + ((size_t)b * 3 + v) * NPTS + n] = acc[v] * (1.f / KNN);
}

// ---------------------------------------------------------------------------
// 3. fp32 -> f16 weight conversion (once per launch; weights stay L2-resident)
// ---------------------------------------------------------------------------
__global__ __launch_bounds__(256)
void cvt_f16_kernel(const float* __restrict__ s, _Float16* __restrict__ d, int n)
{
    const int t = blockIdx.x * 256 + threadIdx.x;
    if (t < n) d[t] = (_Float16)s[t];
}

// ---------------------------------------------------------------------------
// 4. WMMA GEMM:  Y[O x MCOL] (+)= W[O x C] * X[C x MCOL]
//    Block tile 64x128, 8 waves (2x4), wave tile 32x32 = 2x2 WMMA 16x16x32.
//    In this code's k order (k = 16*(lane/16) + s), each lane's A fragment is two contiguous
//    8-half chunks of a single row (kk = 0..15 | 16..31),
//    and each lane's B fragment is 16 contiguous k-values of one column.
//    So: A staged row-major, B staged column-major -> all fragment loads are
//    ds_load_b128, and B staging stores are 16B vector stores.
// ---------------------------------------------------------------------------
#define GBM 64
#define GBN 128
#define GBK 32
#define APAD 8
#define BPAD 8

__device__ inline v16h combine16(v8h lo, v8h hi)
{
    union { v16h v; v8h h[2]; } u;
    u.h[0] = lo; u.h[1] = hi;
    return u.v;
}

template<int C, bool ACC>
__global__ __launch_bounds__(256)
void gemm_wmma_kernel(const _Float16* __restrict__ W, const float* __restrict__ X,
                      float* __restrict__ Y)
{
    __shared__ __align__(16) _Float16 As [GBM][GBK + APAD];   // row-major (o, k)
    __shared__ __align__(16) _Float16 Bst[GBN][GBK + BPAD];   // col-major (n, k)

    const int tid  = threadIdx.x;
    const int lane = tid & 31;
    const int wave = tid >> 5;
    const int wr   = wave >> 2;               // 0..1  (row of wave grid)
    const int wc   = wave & 3;                // 0..3  (col of wave grid)
    const int row0 = blockIdx.y * GBM;
    const int col0 = blockIdx.x * GBN;

    v8f acc[2][2];
    #pragma unroll
    for (int i = 0; i < 2; ++i)
        #pragma unroll
        for (int j = 0; j < 2; ++j)
            #pragma unroll
            for (int e = 0; e < 8; ++e) acc[i][j][e] = 0.f;

    // staging assignments (loop-invariant)
    const int ar = tid >> 2;                  // A: row 0..63
    const int ac = (tid & 3) * 8;             // A: k-chunk {0,8,16,24}
    const int bcol_s = tid & 127;             // B: column 0..127
    const int bkh    = tid >> 7;              // B: k-half 0..1

    // fragment addresses (loop-invariant)
    const int arow = wr * 32 + (lane & 15);
    const int akb  = (lane & 16) ? 16 : 0;    // A k-base per lane half
    const int bcol = wc * 32 + (lane & 15);
    const int bkb  = (lane & 16) ? 16 : 0;    // B k-base per lane half

    for (int k0 = 0; k0 < C; k0 += GBK) {
        // --- stage A (f16 weights): 16B vector load + 16B LDS store ---
        {
            const uint4 v = *reinterpret_cast<const uint4*>(
                W + (row0 + ar) * C + k0 + ac);
            *reinterpret_cast<uint4*>(&As[ar][ac]) = v;
        }
        // --- stage B: one column x 16 k-values per thread (coalesced across
        //     lanes per k), fp32 -> f16, two 16B LDS stores ---
        {
            const float* src = X + (k0 + bkh * 16) * MCOL + col0 + bcol_s;
            v8h p0, p1;
            #pragma unroll
            for (int k = 0; k < 8; ++k) p0[k] = (_Float16)src[k * MCOL];
            #pragma unroll
            for (int k = 0; k < 8; ++k) p1[k] = (_Float16)src[(k + 8) * MCOL];
            *reinterpret_cast<v8h*>(&Bst[bcol_s][bkh * 16 + 0]) = p0;
            *reinterpret_cast<v8h*>(&Bst[bcol_s][bkh * 16 + 8]) = p1;
        }
        __syncthreads();

        // --- fragments: 2x ds_load_b128 each ---
        v16h a[2], bf[2];
        #pragma unroll
        for (int i = 0; i < 2; ++i) {
            const int row = arow + i * 16;
            const v8h lo = *reinterpret_cast<const v8h*>(&As[row][akb]);
            const v8h hi = *reinterpret_cast<const v8h*>(&As[row][akb + 8]);
            a[i] = combine16(lo, hi);
        }
        #pragma unroll
        for (int j = 0; j < 2; ++j) {
            const int col = bcol + j * 16;
            const v8h lo = *reinterpret_cast<const v8h*>(&Bst[col][bkb]);
            const v8h hi = *reinterpret_cast<const v8h*>(&Bst[col][bkb + 8]);
            bf[j] = combine16(lo, hi);
        }

        #pragma unroll
        for (int i = 0; i < 2; ++i)
            #pragma unroll
            for (int j = 0; j < 2; ++j)
                acc[i][j] = __builtin_amdgcn_wmma_f32_16x16x32_f16(
                    false, a[i], false, bf[j], (short)0, acc[i][j], false, false);
        __syncthreads();
    }

    // --- epilogue: C/D layout M = r + 8*(lane>=16), N = lane&15 ---
    const int ncol = col0 + wc * 32 + (lane & 15);
    const int mrow = row0 + wr * 32 + ((lane & 16) ? 8 : 0);
    #pragma unroll
    for (int i = 0; i < 2; ++i) {
        #pragma unroll
        for (int j = 0; j < 2; ++j) {
            float* dst = Y + (mrow + i * 16) * MCOL + ncol + j * 16;
            if constexpr (ACC) {
                float old[8];
                #pragma unroll
                for (int r = 0; r < 8; ++r) old[r] = dst[r * MCOL];
                #pragma unroll
                for (int r = 0; r < 8; ++r) dst[r * MCOL] = old[r] + acc[i][j][r];
            } else {
                #pragma unroll
                for (int r = 0; r < 8; ++r) dst[r * MCOL] = acc[i][j][r];
            }
        }
    }
}

// ---------------------------------------------------------------------------
// 5. VN leaky ReLU across the 3 vector components (columns N apart)
// ---------------------------------------------------------------------------
__global__ __launch_bounds__(256)
void vn_leaky_kernel(const float* __restrict__ Yv, const float* __restrict__ Dv,
                     float* __restrict__ Aout, int O, float slope)
{
    const int BN = BATCH * NPTS;
    const int t  = blockIdx.x * 256 + threadIdx.x;
    if (t >= O * BN) return;
    const int o   = t / BN;
    const int rem = t - o * BN;
    const int b   = rem / NPTS;
    const int n   = rem - b * NPTS;
    const int base = o * MCOL + b * 3 * NPTS + n;

    const float y0 = Yv[base], y1 = Yv[base + NPTS], y2 = Yv[base + 2 * NPTS];
    const float d0 = Dv[base], d1 = Dv[base + NPTS], d2 = Dv[base + 2 * NPTS];
    const float dot = y0 * d0 + y1 * d1 + y2 * d2;
    const float dsq = d0 * d0 + d1 * d1 + d2 * d2;
    const float tt  = dot / (dsq + EPSV);
    const bool  pos = (dot >= 0.f);
    Aout[base]            = slope * y0 + (1.f - slope) * (pos ? y0 : y0 - tt * d0);
    Aout[base + NPTS]     = slope * y1 + (1.f - slope) * (pos ? y1 : y1 - tt * d1);
    Aout[base + 2 * NPTS] = slope * y2 + (1.f - slope) * (pos ? y2 : y2 - tt * d2);
}

// ---------------------------------------------------------------------------
// 6. Mean-pool over n:  P[o*12 + bv] = mean_n S[o, bv*N + n]
// ---------------------------------------------------------------------------
__global__ __launch_bounds__(256)
void pool_kernel(const float* __restrict__ S, float* __restrict__ P)
{
    __shared__ float red[256];
    const int o  = blockIdx.x / (BATCH * 3);
    const int bv = blockIdx.x % (BATCH * 3);
    float s = 0.f;
    for (int n = threadIdx.x; n < NPTS; n += 256)
        s += S[o * MCOL + bv * NPTS + n];
    red[threadIdx.x] = s;
    __syncthreads();
    for (int st = 128; st > 0; st >>= 1) {
        if (threadIdx.x < st) red[threadIdx.x] += red[threadIdx.x + st];
        __syncthreads();
    }
    if (threadIdx.x == 0) P[blockIdx.x] = red[0] * (1.f / NPTS);
}

// ---------------------------------------------------------------------------
// 7. Concat [net ; broadcast(pooled)] -> X256 [256 x MCOL]
// ---------------------------------------------------------------------------
__global__ __launch_bounds__(256)
void concat_kernel(const float* __restrict__ S, const float* __restrict__ P,
                   float* __restrict__ X)
{
    const size_t t     = (size_t)blockIdx.x * 256 + threadIdx.x;
    const size_t total = (size_t)256 * MCOL;
    if (t >= total) return;
    const int r = (int)(t / MCOL);
    const int c = (int)(t - (size_t)r * MCOL);
    X[t] = (r < 128) ? S[r * MCOL + c]
                     : P[(r - 128) * (BATCH * 3) + c / NPTS];
}

// ---------------------------------------------------------------------------
// 8. Final head:  leaky(c, w_actc_dir*c, 0.2) then w_fc_c * (.)  on [128 x 12]
// ---------------------------------------------------------------------------
__global__ __launch_bounds__(128)
void head_kernel(const float* __restrict__ cpre, const float* __restrict__ wdir,
                 const float* __restrict__ wfc, float* __restrict__ out)
{
    __shared__ float cp[128][12];
    __shared__ float al[128][12];
    const int o = threadIdx.x;
    #pragma unroll
    for (int bv = 0; bv < 12; ++bv) cp[o][bv] = cpre[o * 12 + bv];
    __syncthreads();

    float d[12];
    #pragma unroll
    for (int bv = 0; bv < 12; ++bv) d[bv] = 0.f;
    for (int c = 0; c < 128; ++c) {
        const float w = wdir[o * 128 + c];
        #pragma unroll
        for (int bv = 0; bv < 12; ++bv) d[bv] += w * cp[c][bv];
    }
    #pragma unroll
    for (int b = 0; b < BATCH; ++b) {
        const float y0 = cp[o][b * 3 + 0], y1 = cp[o][b * 3 + 1], y2 = cp[o][b * 3 + 2];
        const float e0 = d[b * 3 + 0],     e1 = d[b * 3 + 1],     e2 = d[b * 3 + 2];
        const float dot = y0 * e0 + y1 * e1 + y2 * e2;
        const float dsq = e0 * e0 + e1 * e1 + e2 * e2;
        const float tt  = dot / (dsq + EPSV);
        const bool pos  = (dot >= 0.f);
        const float sl  = 0.2f;
        al[o][b * 3 + 0] = sl * y0 + (1.f - sl) * (pos ? y0 : y0 - tt * e0);
        al[o][b * 3 + 1] = sl * y1 + (1.f - sl) * (pos ? y1 : y1 - tt * e1);
        al[o][b * 3 + 2] = sl * y2 + (1.f - sl) * (pos ? y2 : y2 - tt * e2);
    }
    __syncthreads();

    float r[12];
    #pragma unroll
    for (int bv = 0; bv < 12; ++bv) r[bv] = 0.f;
    for (int c = 0; c < 128; ++c) {
        const float w = wfc[o * 128 + c];
        #pragma unroll
        for (int bv = 0; bv < 12; ++bv) r[bv] += w * al[c][bv];
    }
    #pragma unroll
    for (int bv = 0; bv < 12; ++bv) {
        const int b = bv / 3, v = bv % 3;
        out[((size_t)b * 128 + o) * 3 + v] = r[bv];     // [B, 128, 3]
    }
}

// ---------------------------------------------------------------------------
// Host-side launcher
// ---------------------------------------------------------------------------
extern "C" void kernel_launch(void* const* d_in, const int* in_sizes, int n_in,
                              void* d_out, int out_size, void* d_ws, size_t ws_size,
                              hipStream_t stream)
{
    (void)in_sizes; (void)n_in; (void)out_size; (void)ws_size;
    const float* p       = (const float*)d_in[0];
    const float* w_cf    = (const float*)d_in[1];
    const float* w_cd    = (const float*)d_in[2];
    const float* w_fcpos = (const float*)d_in[3];
    const float* bd0     = (const float*)d_in[4];
    const float* bf0     = (const float*)d_in[5];
    const float* bd1     = (const float*)d_in[6];
    const float* bf1     = (const float*)d_in[7];
    const float* bsc     = (const float*)d_in[8];
    const float* w_actd  = (const float*)d_in[9];
    const float* w_fcc   = (const float*)d_in[10];

    char* ws = (char*)d_ws;
    size_t off = 0;
    auto take = [&](size_t bytes) -> char* {
        char* r = ws + off;
        off = (off + bytes + 255) & ~(size_t)255;
        return r;
    };

    int*      idxb    = (int*)take(sizeof(int) * (size_t)BATCH * NPTS * KNN);
    _Float16* h_fcpos = (_Float16*)take(2 * (size_t)256 * 128);
    _Float16* h_d0    = (_Float16*)take(2 * (size_t)5 * 256 * 256);
    _Float16* h_f0    = (_Float16*)take(2 * (size_t)5 * 128 * 256);
    _Float16* h_d1    = (_Float16*)take(2 * (size_t)5 * 128 * 128);
    _Float16* h_f1    = (_Float16*)take(2 * (size_t)5 * 128 * 128);
    _Float16* h_sc    = (_Float16*)take(2 * (size_t)5 * 128 * 256);
    float*    X0      = (float*)take(4 * (size_t)128 * MCOL);
    float*    X256    = (float*)take(4 * (size_t)256 * MCOL);
    float*    Dbuf    = (float*)take(4 * (size_t)256 * MCOL);
    float*    Abuf    = (float*)take(4 * (size_t)256 * MCOL);
    float*    Tnet    = (float*)take(4 * (size_t)128 * MCOL);
    float*    Bout    = (float*)take(4 * (size_t)128 * MCOL);
    float*    pooled  = (float*)take(4 * (size_t)128 * 12);
    float*    cpre    = (float*)take(4 * (size_t)128 * 12);

    // 1. kNN indices
    knn_kernel<<<dim3(NPTS / 8, BATCH), 256, 0, stream>>>(p, idxb);

    // 2. weights -> f16
    auto cvt = [&](const float* s, _Float16* dst, int n) {
        cvt_f16_kernel<<<(n + 255) / 256, 256, 0, stream>>>(s, dst, n);
    };
    cvt(w_fcpos, h_fcpos, 256 * 128);
    cvt(bd0, h_d0, 5 * 256 * 256);
    cvt(bf0, h_f0, 5 * 128 * 256);
    cvt(bd1, h_d1, 5 * 128 * 128);
    cvt(bf1, h_f1, 5 * 128 * 128);
    cvt(bsc, h_sc, 5 * 128 * 256);

    // 3. edge features + conv + k-pool -> X0 [128 x M]
    edge_conv_kernel<<<dim3(NPTS, BATCH), 128, 0, stream>>>(p, idxb, w_cf, w_cd, X0);

    auto gemm = [&](const _Float16* W, const float* X, float* Y, int O, int C, int acc) {
        dim3 g(MCOL / GBN, O / GBM);
        if (C == 128) {
            if (acc) gemm_wmma_kernel<128, true ><<<g, 256, 0, stream>>>(W, X, Y);
            else     gemm_wmma_kernel<128, false><<<g, 256, 0, stream>>>(W, X, Y);
        } else {
            if (acc) gemm_wmma_kernel<256, true ><<<g, 256, 0, stream>>>(W, X, Y);
            else     gemm_wmma_kernel<256, false><<<g, 256, 0, stream>>>(W, X, Y);
        }
    };

    // 4. fc_pos: [256x128] x [128xM] -> X256
    gemm(h_fcpos, X0, X256, 256, 128, 0);

    // 5. five resnet blocks
    for (int i = 0; i < 5; ++i) {
        if (i > 0) {
            pool_kernel<<<128 * 12, 256, 0, stream>>>(Bout, pooled);
            concat_kernel<<<(int)(((size_t)256 * MCOL + 255) / 256), 256, 0, stream>>>(
                Bout, pooled, X256);
        }
        gemm(h_d0 + (size_t)i * 256 * 256, X256, Dbuf, 256, 256, 0);
        vn_leaky_kernel<<<(256 * BATCH * NPTS) / 256, 256, 0, stream>>>(X256, Dbuf, Abuf, 256, 0.0f);
        gemm(h_f0 + (size_t)i * 128 * 256, Abuf, Tnet, 128, 256, 0);
        gemm(h_d1 + (size_t)i * 128 * 128, Tnet, Dbuf, 128, 128, 0);
        vn_leaky_kernel<<<(128 * BATCH * NPTS) / 256, 256, 0, stream>>>(Tnet, Dbuf, Abuf, 128, 0.0f);
        gemm(h_sc + (size_t)i * 128 * 256, X256, Bout, 128, 256, 0);   // Y = wsc * x
        gemm(h_f1 + (size_t)i * 128 * 128, Abuf, Bout, 128, 128, 1);   // Y += wf1 * a
    }

    // 6. global pool + head
    pool_kernel<<<128 * 12, 256, 0, stream>>>(Bout, cpre);
    head_kernel<<<1, 128, 0, stream>>>(cpre, w_actd, w_fcc, (float*)d_out);
}
